// CausalAttention_2319282340620
// MI455X (gfx1250) — matrix operand order repaired and checked by reading, not compiled, not run on hardware
//
#include <hip/hip_runtime.h>

#define DEV_INLINE __device__ __forceinline__

typedef __attribute__((ext_vector_type(16))) _Float16 v16h;
typedef __attribute__((ext_vector_type(8)))  _Float16 v8h;
typedef __attribute__((ext_vector_type(4)))  _Float16 v4h;
typedef __attribute__((ext_vector_type(8)))  float    v8f;

union AFrag { v16h v; v8h h8[2]; };
union CFrag { v8f  v; float f[8]; };

static constexpr int Bc = 2, Tc = 2048, Cc = 2048, Hc = 16, Dc = 128;
static constexpr int Mrows = Bc * Tc; // 4096

DEV_INLINE v8f wmma_f16(v16h a, v16h b, v8f c) {
  // D = A(16x32 f16) * B(32x16 f16) + C(16x16 f32)
  return __builtin_amdgcn_wmma_f32_16x16x32_f16(false, a, false, b,
                                                (short)0, c, false, false);
}

// Async global->LDS copy, 16B per lane. Tracked by ASYNCcnt (not DScnt/LOADcnt).
// Generic pointers to __shared__ carry the LDS byte offset in their low 32 bits
// (per the ISA's flat LDS aperture mapping).
DEV_INLINE void async_ld_b128(const _Float16* gsrc, _Float16* ldst) {
  unsigned loff = (unsigned)(uintptr_t)ldst;
  unsigned long long ga = (unsigned long long)(uintptr_t)gsrc;
  asm volatile("global_load_async_to_lds_b128 %0, %1, off"
               :: "v"(loff), "v"(ga) : "memory");
}
DEV_INLINE void wait_async0() {
  asm volatile("s_wait_asynccnt 0x0" ::: "memory");
}

// ---------------------------------------------------------------------------
// Kernel 0: f32 -> f16 conversion of activations
// ---------------------------------------------------------------------------
__global__ void cvt_f32_f16(const float* __restrict__ x, _Float16* __restrict__ y) {
  size_t i = ((size_t)blockIdx.x * 256 + threadIdx.x) * 4;
  float4 f = *(const float4*)(x + i);
  v4h o;
  o[0] = (_Float16)f.x; o[1] = (_Float16)f.y;
  o[2] = (_Float16)f.z; o[3] = (_Float16)f.w;
  *(v4h*)(y + i) = o;
}

// ---------------------------------------------------------------------------
// Kernel 1: GEMM  out[M,N] = A_f16[M,K] * W_f32[K,N] + bias
//   block = 128 threads (4 waves). Each wave: 16 rows x 64 cols.
//   B tile (32xK x 64xN) staged f32->f16 into LDS transposed [n][k] so that a
//   B fragment is one contiguous 32B LDS read per lane.
//   headShuffle=1: store f16 into [B,H,T,D] (QKV). headShuffle=0: f32 [M,N].
// ---------------------------------------------------------------------------
__global__ void gemm16_wmma(const _Float16* __restrict__ A,
                            const float* __restrict__ W,
                            const float* __restrict__ bias,
                            _Float16* __restrict__ outHead,
                            float* __restrict__ outF32,
                            int headShuffle) {
  __shared__ __align__(32) _Float16 ldsBt[64 * 32]; // [n_local][k_local]

  const int tid  = threadIdx.x;
  const int wv   = tid >> 5;
  const int lane = tid & 31;
  const int grp  = lane >> 4;   // half-group (0/1)
  const int lm   = lane & 15;
  const int m0   = blockIdx.y * 64 + wv * 16;
  const int n0   = blockIdx.x * 64;

  CFrag acc[4];
#pragma unroll
  for (int i = 0; i < 4; i++)
#pragma unroll
    for (int e = 0; e < 8; e++) acc[i].f[e] = 0.0f;

  const int ks = tid >> 2; // staging: k row 0..31
  const int nq = tid & 3;  // staging: 16-col group 0..3

  for (int k0 = 0; k0 < Cc; k0 += 32) {
    __syncthreads();
    { // stage W[k0+ks][n0 + nq*16 .. +15] -> ldsBt[n][k] (f16, transposed)
      const float4* src = (const float4*)(W + (size_t)(k0 + ks) * Cc + n0 + nq * 16);
#pragma unroll
      for (int i = 0; i < 4; i++) {
        float4 f = src[i];
        _Float16* d = &ldsBt[(nq * 16 + i * 4) * 32 + ks];
        d[0 * 32] = (_Float16)f.x;
        d[1 * 32] = (_Float16)f.y;
        d[2 * 32] = (_Float16)f.z;
        d[3 * 32] = (_Float16)f.w;
      }
    }
    __syncthreads();

    // A fragment: 16x32 f16, lane row = lm, K = grp*16+i
    AFrag af;
    const _Float16* arow = A + (size_t)(m0 + lm) * Cc + k0 + grp * 16;
    af.h8[0] = *(const v8h*)(arow);
    af.h8[1] = *(const v8h*)(arow + 8);

#pragma unroll
    for (int nt = 0; nt < 4; nt++) {
      // B fragment: element j = W[k0 + grp*16 + j][n0 + nt*16 + lm]
      v16h bf = *(const v16h*)&ldsBt[(nt * 16 + lm) * 32 + grp * 16];
      acc[nt].v = wmma_f16(af.v, bf, acc[nt].v);
    }
  }

  // epilogue: C layout: vgpr r -> M = r + grp*8, N = lm
#pragma unroll
  for (int nt = 0; nt < 4; nt++) {
    const int cg = n0 + nt * 16 + lm;
    const float bv = bias[cg];
#pragma unroll
    for (int r = 0; r < 8; r++) {
      const int mg = m0 + r + grp * 8;
      const float val = acc[nt].f[r] + bv;
      if (headShuffle) {
        const int bb = mg >> 11, t = mg & (Tc - 1);
        const int hh = cg >> 7, d = cg & (Dc - 1);
        outHead[(((size_t)(bb * Hc + hh) * Tc + t) << 7) + d] = (_Float16)val;
      } else {
        outF32[(size_t)mg * Cc + cg] = val;
      }
    }
  }
}

// ---------------------------------------------------------------------------
// Kernel 2: flash attention with ALiBi + causal mask.
//   block = 128 threads (4 waves), one 64-query band of one (b,h).
//   Per 32-key chunk: K chunk [32][128] staged by async global->LDS DMA
//   (ASYNCcnt path), V^T chunk [128][32] staged through VALU (transpose).
//   Per wave: 8 WMMA for S, online softmax, 8 WMMA for P*V.
// ---------------------------------------------------------------------------
__global__ void attn_wmma(const _Float16* __restrict__ Q,
                          const _Float16* __restrict__ Kg,
                          const _Float16* __restrict__ Vg,
                          _Float16* __restrict__ O16) {
  __shared__ __align__(32) _Float16 ldsK[32 * 128];    // [k][d]
  __shared__ __align__(32) _Float16 ldsV[128 * 32];    // [d][k] transposed
  __shared__ __align__(32) _Float16 ldsP[4][16 * 32];  // per-wave P tile [q][k]

  const int tid  = threadIdx.x;
  const int wv   = tid >> 5;
  const int lane = tid & 31;
  const int grp  = lane >> 4;
  const int lm   = lane & 15;

  const int qband = (blockIdx.x & 31) * 64; // 32 bands of 64 queries
  const int bh    = blockIdx.x >> 5;
  const int hh    = bh & (Hc - 1);
  const int bb    = bh >> 4;

  const int q0 = qband + wv * 16;
  const size_t base = (size_t)(bb * Hc + hh) * Tc * Dc;
  const _Float16* Qb = Q + base;
  const _Float16* Kb = Kg + base;
  const _Float16* Vb = Vg + base;

  // Q fragments, 4 d-slices of 32 (held in registers for the whole kernel)
  AFrag qf[4];
  {
    const _Float16* qrow = Qb + (size_t)(q0 + lm) * Dc;
#pragma unroll
    for (int s = 0; s < 4; s++) {
      qf[s].h8[0] = *(const v8h*)(qrow + s * 32 + grp * 16);
      qf[s].h8[1] = *(const v8h*)(qrow + s * 32 + grp * 16 + 8);
    }
  }

  float mrow[8], lrow[8];
  CFrag od[8];
#pragma unroll
  for (int r = 0; r < 8; r++) { mrow[r] = -1e30f; lrow[r] = 0.0f; }
#pragma unroll
  for (int dt = 0; dt < 8; dt++)
#pragma unroll
    for (int e = 0; e < 8; e++) od[dt].f[e] = 0.0f;

  const float slope = exp2f(-0.5f * (float)(hh + 1)); // ALiBi slopes for H=16
  const float scl   = 0.08838834764831845f;           // 1/sqrt(128)

  const int ks = tid >> 2; // staging: key row 0..31
  const int qd = tid & 3;  // staging: 32-elem d group

  const int nchunks = (qband >> 5) + 2; // covers keys 0 .. qband+63
  for (int c = 0; c < nchunks; c++) {
    const int k0 = c * 32;
    __syncthreads();
    { // stage K via async global->LDS DMA; V transposed through VALU.
      int krow = k0 + ks; if (krow > Tc - 1) krow = Tc - 1;
      const _Float16* ksrc = Kb + (size_t)krow * Dc + qd * 32;
      _Float16* kdst = &ldsK[ks * 128 + qd * 32];
#pragma unroll
      for (int i = 0; i < 4; i++)
        async_ld_b128(ksrc + i * 8, kdst + i * 8);

      const v8h* vsrc = (const v8h*)(Vb + (size_t)krow * Dc + qd * 32);
#pragma unroll
      for (int i = 0; i < 4; i++) {
        v8h vvv = vsrc[i];
#pragma unroll
        for (int e = 0; e < 8; e++)
          ldsV[(qd * 32 + i * 8 + e) * 32 + ks] = vvv[e];
      }
      wait_async0(); // this thread's async LDS writes complete
    }
    __syncthreads();   // all threads' staging visible

    // S = Q * K^T  (two 16-key sub-tiles)
    CFrag s0, s1;
#pragma unroll
    for (int e = 0; e < 8; e++) { s0.f[e] = 0.0f; s1.f[e] = 0.0f; }
#pragma unroll
    for (int s = 0; s < 4; s++) {
      v16h kf0 = *(const v16h*)&ldsK[(lm) * 128 + s * 32 + grp * 16];
      s0.v = wmma_f16(qf[s].v, kf0, s0.v);
      v16h kf1 = *(const v16h*)&ldsK[(16 + lm) * 128 + s * 32 + grp * 16];
      s1.v = wmma_f16(qf[s].v, kf1, s1.v);
    }

    // scale + ALiBi + causal mask
    const int kg0 = k0 + lm, kg1 = k0 + 16 + lm;
    const float cb0 = slope * (float)(kg0 - (Tc - 1));
    const float cb1 = slope * (float)(kg1 - (Tc - 1));
    float rmax[8];
#pragma unroll
    for (int r = 0; r < 8; r++) {
      const int qg = q0 + r + grp * 8;
      float a0 = s0.f[r] * scl + cb0; if (kg0 > qg) a0 = -1e30f;
      float a1 = s1.f[r] * scl + cb1; if (kg1 > qg) a1 = -1e30f;
      s0.f[r] = a0; s1.f[r] = a1;
      rmax[r] = fmaxf(a0, a1);
    }
    // row max across the 16 lanes of each half-group
#pragma unroll
    for (int msk = 1; msk < 16; msk <<= 1)
#pragma unroll
      for (int r = 0; r < 8; r++)
        rmax[r] = fmaxf(rmax[r], __shfl_xor(rmax[r], msk, 32));

    float psum[8], sc[8];
#pragma unroll
    for (int r = 0; r < 8; r++) {
      const float mnew = fmaxf(mrow[r], rmax[r]);
      sc[r] = __expf(mrow[r] - mnew);
      mrow[r] = mnew;
      const float p0 = __expf(s0.f[r] - mnew);
      const float p1 = __expf(s1.f[r] - mnew);
      psum[r] = p0 + p1;
      const int rowm = r + grp * 8;
      ldsP[wv][rowm * 32 + lm]      = (_Float16)p0;
      ldsP[wv][rowm * 32 + 16 + lm] = (_Float16)p1;
    }
#pragma unroll
    for (int msk = 1; msk < 16; msk <<= 1)
#pragma unroll
      for (int r = 0; r < 8; r++)
        psum[r] += __shfl_xor(psum[r], msk, 32);
#pragma unroll
    for (int r = 0; r < 8; r++) lrow[r] = lrow[r] * sc[r] + psum[r];
#pragma unroll
    for (int dt = 0; dt < 8; dt++)
#pragma unroll
      for (int r = 0; r < 8; r++) od[dt].f[r] *= sc[r];

    __builtin_amdgcn_wave_barrier(); // LDS is in-order per wave; pin compiler order

    // P as A fragment (in-wave LDS transpose), then O += P * V
    AFrag pf;
    pf.h8[0] = *(const v8h*)&ldsP[wv][lm * 32 + grp * 16];
    pf.h8[1] = *(const v8h*)&ldsP[wv][lm * 32 + grp * 16 + 8];
#pragma unroll
    for (int dt = 0; dt < 8; dt++) {
      v16h vf = *(const v16h*)&ldsV[(dt * 16 + lm) * 32 + grp * 16];
      od[dt].v = wmma_f16(pf.v, vf, od[dt].v);
    }
  }

  // normalize and store O (f16, [B,T,C] with c = h*128 + d)
#pragma unroll
  for (int r = 0; r < 8; r++) {
    const int t = q0 + r + grp * 8;
    const float inv = 1.0f / lrow[r];
    const size_t rowoff = (size_t)(bb * Tc + t) * Cc + hh * Dc;
#pragma unroll
    for (int dt = 0; dt < 8; dt++)
      O16[rowoff + dt * 16 + lm] = (_Float16)(od[dt].f[r] * inv);
  }
}

// ---------------------------------------------------------------------------
extern "C" void kernel_launch(void* const* d_in, const int* in_sizes, int n_in,
                              void* d_out, int out_size, void* d_ws, size_t ws_size,
                              hipStream_t stream) {
  (void)in_sizes; (void)n_in; (void)out_size; (void)ws_size;
  const float* x  = (const float*)d_in[0];
  const float* Wq = (const float*)d_in[1];
  const float* bq = (const float*)d_in[2];
  const float* Wk = (const float*)d_in[3];
  const float* bk = (const float*)d_in[4];
  const float* Wv = (const float*)d_in[5];
  const float* bv = (const float*)d_in[6];
  const float* Wo = (const float*)d_in[7];
  const float* bo = (const float*)d_in[8];
  float* out = (float*)d_out;

  const size_t elems = (size_t)Mrows * Cc; // 8,388,608 per buffer
  _Float16* x16 = (_Float16*)d_ws;
  _Float16* Qb  = x16 + elems;
  _Float16* Kb  = Qb + elems;
  _Float16* Vb  = Kb + elems;
  _Float16* O16 = Vb + elems;  // total 80 MB of workspace

  cvt_f32_f16<<<(int)(elems / 1024), 256, 0, stream>>>(x, x16);

  dim3 gg(Cc / 64, Mrows / 64); // (32, 64)
  gemm16_wmma<<<gg, 128, 0, stream>>>(x16, Wq, bq, Qb, nullptr, 1);
  gemm16_wmma<<<gg, 128, 0, stream>>>(x16, Wk, bk, Kb, nullptr, 1);
  gemm16_wmma<<<gg, 128, 0, stream>>>(x16, Wv, bv, Vb, nullptr, 1);

  attn_wmma<<<Bc * Hc * (Tc / 64), 128, 0, stream>>>(Qb, Kb, Vb, O16);

  gemm16_wmma<<<gg, 128, 0, stream>>>(O16, Wo, bo, nullptr, out, 0);
}
